// GNNLF_87445534146858
// MI455X (gfx1250) — compile-verified, matrix operand order checked by reading, not run on hardware
//
#include <hip/hip_runtime.h>
#include <hip/hip_bf16.h>
#include <math.h>

// ---------------------------------------------------------------------------
// GNN-LF forward on MI455X (gfx1250), wave32 + WMMA f16 16x16x32.
// Pair-GEMMs (ef@Wef, dirf@Wd1, h1@Wd2) dominate (~55 GFLOP) and run on the
// matrix pipes with fp16-in / fp32-acc. Weights are pre-packed once per launch
// into per-lane WMMA fragment order (f16), so the GEMM inner loop is just
// b128 loads + back-to-back WMMAs (4 accumulators per wave reuse one A frag).
// Streamed intermediates ~0.5 GB, trivial at 23.3 TB/s.
// ---------------------------------------------------------------------------

typedef _Float16 half_t;
typedef __attribute__((ext_vector_type(16))) _Float16 v16h;
typedef __attribute__((ext_vector_type(8)))  _Float16 v8h;
typedef __attribute__((ext_vector_type(8)))  float    v8f;

#define BB   64
#define NA   64
#define HID  128
#define EF   32
#define DIR  256
#define ROWS (BB * NA)          // 4096 atoms
#define NP   (BB * NA * NA)     // 262144 pairs
#define LN_EPS 1e-5f
#define CUTOFF 10.0f
#define RBF_GAMMA 10.0f
#define PI_F 3.14159265358979f

// silu with raw v_rcp_f32 (no IEEE div fixup chain)
__device__ __forceinline__ float silu_f(float x) {
    return x * __builtin_amdgcn_rcpf(1.0f + __expf(-x));
}

// ---------------------------------------------------------------------------
// pack_w: convert W[K,NOUT] (f32, row-major) into WMMA B-fragment order (f16):
//   dst[((chunk*(NOUT/16)+tile)*32 + lane)*16 + e]  with
//   k = chunk*32 + (lane>>4)*8 + (e<8 ? e : e+8), col = tile*16 + (lane&15)
// so a lane's 16 halves are two contiguous 16B runs -> b128 loads in the GEMM.
// ---------------------------------------------------------------------------
__global__ __launch_bounds__(256) void pack_w_k(half_t* __restrict__ dst,
                                                const float* __restrict__ W,
                                                int K, int NOUT) {
    const int idx = blockIdx.x * 256 + threadIdx.x;
    if (idx >= K * NOUT) return;
    const int e    = idx & 15;
    const int lane = (idx >> 4) & 31;
    const int blk  = idx >> 9;
    const int tpc  = NOUT >> 4;                  // tiles per chunk
    const int tile = blk % tpc;
    const int chunk = blk / tpc;
    const int k   = chunk * 32 + ((lane >> 4) << 3) + (e < 8 ? e : e + 8);
    const int col = tile * 16 + (lane & 15);
    dst[idx] = (half_t)W[(size_t)k * NOUT + col];
}

// ---------------------------------------------------------------------------
// embed: s = emb_w[z] (+f16 shadow); sn = LayerNorm(emb2_w[z]).
// ---------------------------------------------------------------------------
__global__ __launch_bounds__(128) void embed_k(float* __restrict__ s,
                                               half_t* __restrict__ s_h,
                                               float* __restrict__ sn,
                                               const int* __restrict__ z,
                                               const float* __restrict__ emb_w,
                                               const float* __restrict__ emb2_w) {
    const int row = blockIdx.x;     // b*NA + i
    const int h   = threadIdx.x;
    const int zi  = z[row];
    const float sv = emb_w[(size_t)zi * HID + h];
    s[(size_t)row * HID + h]   = sv;
    s_h[(size_t)row * HID + h] = (half_t)sv;

    float x = emb2_w[(size_t)zi * HID + h];
    __shared__ float red[HID];
    red[h] = x; __syncthreads();
    for (int st = 64; st > 0; st >>= 1) { if (h < st) red[h] += red[h + st]; __syncthreads(); }
    const float mean = red[0] * (1.0f / HID);
    __syncthreads();
    const float dm = x - mean;
    red[h] = dm * dm; __syncthreads();
    for (int st = 64; st > 0; st >>= 1) { if (h < st) red[h] += red[h + st]; __syncthreads(); }
    const float var = red[0] * (1.0f / HID);
    sn[(size_t)row * HID + h] = dm * __frsqrt_rn(var + LN_EPS);
}

// ---------------------------------------------------------------------------
// ef_mask: fused geometry + RBF + (ef @ Wef + bef) via WMMA + silu*ea epilogue.
// Block = 64 thr = 2 waves; each wave owns a 16x64 tile (4 WMMAs reusing the
// register-generated RBF A fragment).  Wef pre-packed (K=32: single chunk).
// ---------------------------------------------------------------------------
__global__ __launch_bounds__(64) void ef_mask_k(const int*    __restrict__ z,
                                                const float*  __restrict__ pos,
                                                const half_t* __restrict__ Wef_p,
                                                const float*  __restrict__ bef,
                                                float* __restrict__ mask,
                                                float* __restrict__ ea_out,
                                                float* __restrict__ ev_out) {
    const int lane = threadIdx.x & 31;
    const int wave = threadIdx.x >> 5;          // column group 0..1 (64 cols each)
    const int rowTile = blockIdx.x;             // 16 pairs
    const int col0 = wave * 64 + (lane & 15);
    const int r    = rowTile * 16 + (lane & 15); // pair index for this lane
    const int b = r >> 12, i = (r >> 6) & 63, j = r & 63;

    // geometry for this lane's pair row
    const float* pi = pos + (size_t)(b * NA + i) * 3;
    const float* pj = pos + (size_t)(b * NA + j) * 3;
    const float dx = pj[0] - pi[0], dy = pj[1] - pi[1], dz = pj[2] - pi[2];
    const float d  = sqrtf(fmaxf(dx * dx + dy * dy + dz * dz, 1e-12f));
    const bool pair = (z[b * NA + i] != 0) && (z[b * NA + j] != 0) && (i != j);
    const float eav = (pair && d < CUTOFF)
                          ? 0.5f * (__cosf(PI_F * d * (1.0f / CUTOFF)) + 1.0f) : 0.0f;
    const float inv = pair ? __builtin_amdgcn_rcpf(d) : 0.0f;
    const float evx = dx * inv, evy = dy * inv, evz = dz * inv;

    // A (RBF feats, 16x32 f16): lane half covers K {0..7,16..23}/{8..15,24..31}
    const int khalf = (lane >> 4) * 8;
    v16h a;
#pragma unroll
    for (int e = 0; e < 16; ++e) {
        const int k = khalf + (e < 8 ? e : e + 8);
        const float ck = (float)k * (CUTOFF / (EF - 1));
        const float t  = d - ck;
        a[e] = (half_t)__expf(-RBF_GAMMA * t * t);
    }

    v8f acc[4];
#pragma unroll
    for (int t = 0; t < 4; ++t) {
        const int tileIdx = wave * 4 + t;
        const float binit = bef[col0 + t * 16];
#pragma unroll
        for (int rv = 0; rv < 8; ++rv) acc[t][rv] = binit;
        const half_t* bp = Wef_p + ((size_t)tileIdx << 9) + lane * 16;
        const v8h b0 = *(const v8h*)bp;
        const v8h b1 = *(const v8h*)(bp + 8);
        v16h bm;
#pragma unroll
        for (int e = 0; e < 8; ++e) { bm[e] = b0[e]; bm[8 + e] = b1[e]; }
        acc[t] = __builtin_amdgcn_wmma_f32_16x16x32_f16(false, a, false, bm,
                                                        (short)0, acc[t], false, false);
    }

    // epilogue: mask = silu(x) * ea (ea fetched from the lane owning that row)
    const int hi = (lane >> 4) * 8;
#pragma unroll
    for (int t = 0; t < 4; ++t) {
        const int col = col0 + t * 16;
#pragma unroll
        for (int rv = 0; rv < 8; ++rv) {
            const int mloc = rv + hi;                       // local row 0..15
            const float ea_m = __shfl(eav, mloc, 32);
            mask[((size_t)(rowTile * 16 + mloc)) * HID + col] = silu_f(acc[t][rv]) * ea_m;
        }
    }
    // side outputs, once per pair
    if (wave == 0 && lane < 16) {
        ea_out[r] = eav;
        ev_out[(size_t)r * 3 + 0] = evx;
        ev_out[(size_t)r * 3 + 1] = evy;
        ev_out[(size_t)r * 3 + 2] = evz;
    }
}

// ---------------------------------------------------------------------------
// Generic WMMA GEMM:  out = act( A[M,K](f16) @ Wp(packed f16) + bias )
//   K,NOUT : compile-time -> fully unrolled, straight-line WMMAs
//   SILU   : apply silu in epilogue
//   OMODE  : 0 = store f32, 1 = store f16, 2 = multiply into existing f32
// Each wave computes a 16x64 tile (4 accumulators reuse one A fragment).
// Inner loop per 32-K chunk: 2 A-loads + 8 packed-B loads (b128) + 4 WMMAs.
// ---------------------------------------------------------------------------
template <int K, int NOUT, bool SILU, int OMODE>
__global__ __launch_bounds__(256) void gemm_wmma_k(const half_t* __restrict__ A,
                                                   const half_t* __restrict__ Wp,
                                                   const float*  __restrict__ bias,
                                                   void* __restrict__ outp,
                                                   int M) {
    constexpr int COLG = NOUT / 64;             // column groups of 64
    constexpr int TPC  = NOUT / 16;             // tiles per K-chunk in packed W
    const int lane = threadIdx.x & 31;
    const int wv   = blockIdx.x * 8 + (threadIdx.x >> 5);
    const int rowTile  = wv / COLG;
    const int colGroup = wv % COLG;
    if (rowTile * 16 >= M) return;              // wave-uniform guard

    const int col0  = colGroup * 64 + (lane & 15);
    const int arow  = rowTile * 16 + (lane & 15);
    const int khalf = (lane >> 4) * 8;

    v8f acc[4];
#pragma unroll
    for (int t = 0; t < 4; ++t) {
        const float binit = bias ? bias[col0 + t * 16] : 0.0f;
#pragma unroll
        for (int rv = 0; rv < 8; ++rv) acc[t][rv] = binit;
    }

    const half_t* Arow = A + (size_t)arow * K;
#pragma unroll
    for (int kc = 0; kc < K; kc += 32) {
        __builtin_prefetch(&Arow[kc + 32], 0, 1);   // global_prefetch_b8
        const v8h a0 = *(const v8h*)(Arow + kc + khalf);
        const v8h a1 = *(const v8h*)(Arow + kc + khalf + 16);
        v16h a;
#pragma unroll
        for (int e = 0; e < 8; ++e) { a[e] = a0[e]; a[8 + e] = a1[e]; }
#pragma unroll
        for (int t = 0; t < 4; ++t) {
            const int tileIdx = colGroup * 4 + t;
            const half_t* bp = Wp + ((((size_t)(kc >> 5)) * TPC + tileIdx) << 9) + lane * 16;
            const v8h b0 = *(const v8h*)bp;
            const v8h b1 = *(const v8h*)(bp + 8);
            v16h bm;
#pragma unroll
            for (int e = 0; e < 8; ++e) { bm[e] = b0[e]; bm[8 + e] = b1[e]; }
            acc[t] = __builtin_amdgcn_wmma_f32_16x16x32_f16(false, a, false, bm,
                                                            (short)0, acc[t], false, false);
        }
    }

    const int hi = (lane >> 4) * 8;
#pragma unroll
    for (int t = 0; t < 4; ++t) {
        const int col = col0 + t * 16;
#pragma unroll
        for (int rv = 0; rv < 8; ++rv) {
            const int m = rowTile * 16 + rv + hi;
            float x = acc[t][rv];
            if (SILU) x = silu_f(x);
            const size_t idx = (size_t)m * NOUT + col;
            if (OMODE == 0)      ((float*)outp)[idx]  = x;
            else if (OMODE == 1) ((half_t*)outp)[idx] = (half_t)x;
            else                 ((float*)outp)[idx] *= x;
        }
    }
}

// ---------------------------------------------------------------------------
// Row LayerNorm (+ optional silu), one block (128 thr) per row.
// ---------------------------------------------------------------------------
template <bool SILU>
__global__ __launch_bounds__(128) void ln_k(const float* __restrict__ in,
                                            float* __restrict__ out) {
    const int row = blockIdx.x;
    const int h   = threadIdx.x;
    const float x = in[(size_t)row * HID + h];
    __shared__ float red[HID];
    red[h] = x; __syncthreads();
    for (int st = 64; st > 0; st >>= 1) { if (h < st) red[h] += red[h + st]; __syncthreads(); }
    const float mean = red[0] * (1.0f / HID);
    __syncthreads();
    const float dm = x - mean;
    red[h] = dm * dm; __syncthreads();
    for (int st = 64; st > 0; st >>= 1) { if (h < st) red[h] += red[h + st]; __syncthreads(); }
    const float var = red[0] * (1.0f / HID);
    float y = dm * __frsqrt_rn(var + LN_EPS);
    if (SILU) y = silu_f(y);
    out[(size_t)row * HID + h] = y;
}

// ---------------------------------------------------------------------------
// agg: s[b,i,h] += sum_j mask[b,i,j,h] * src[b,j,h]  (+f16 shadow of s).
// ---------------------------------------------------------------------------
__global__ __launch_bounds__(128) void agg_k(float* __restrict__ s,
                                             half_t* __restrict__ s_h,
                                             const float* __restrict__ mask,
                                             const float* __restrict__ src) {
    const int row = blockIdx.x;     // b*NA + i
    const int h   = threadIdx.x;
    const int b   = row >> 6;
    float acc = s[(size_t)row * HID + h];
    const float* mrow = mask + (size_t)row * NA * HID + h;
    const float* sb   = src + (size_t)b * NA * HID + h;
#pragma unroll 4
    for (int j = 0; j < NA; ++j)
        acc += mrow[(size_t)j * HID] * sb[(size_t)j * HID];
    s[(size_t)row * HID + h]   = acc;
    s_h[(size_t)row * HID + h] = (half_t)acc;
}

// ---------------------------------------------------------------------------
// cfconv_v: v[b,i,c,h] = sum_j mask[b,i,j,h] * t[b,j,h] * ev[b,i,j,c]
// (writes f32 for dirfeat + f16 shadow for the Wq/Wk GEMMs)
// ---------------------------------------------------------------------------
__global__ __launch_bounds__(128) void cfconv_v_k(float* __restrict__ v,
                                                  half_t* __restrict__ v_h,
                                                  const float* __restrict__ mask,
                                                  const float* __restrict__ t,
                                                  const float* __restrict__ ev) {
    const int row = blockIdx.x;     // b*NA + i
    const int h   = threadIdx.x;
    const int b   = row >> 6;
    float a0 = 0.f, a1 = 0.f, a2 = 0.f;
    for (int j = 0; j < NA; ++j) {
        const size_t p = (size_t)row * NA + j;
        const float mt = mask[p * HID + h] * t[((size_t)b * NA + j) * HID + h];
        a0 += mt * ev[p * 3 + 0];
        a1 += mt * ev[p * 3 + 1];
        a2 += mt * ev[p * 3 + 2];
    }
    const size_t base = ((size_t)row * 3) * HID + h;
    v[base]           = a0;
    v[base + HID]     = a1;
    v[base + 2 * HID] = a2;
    v_h[base]           = (half_t)a0;
    v_h[base + HID]     = (half_t)a1;
    v_h[base + 2 * HID] = (half_t)a2;
}

// ---------------------------------------------------------------------------
// dirfeat: dirf[b,i,j,:] = cat(dir2, dir3) * ea, stored f16 for the big GEMM.
// vq/vk arrive as f16 (GEMM outputs), widened in-register.
// ---------------------------------------------------------------------------
__global__ __launch_bounds__(128) void dirfeat_k(half_t* __restrict__ dirf,
                                                 const float*  __restrict__ v,
                                                 const half_t* __restrict__ vq,
                                                 const half_t* __restrict__ vk,
                                                 const float*  __restrict__ ev,
                                                 const float*  __restrict__ ea) {
    const int r = blockIdx.x;       // pair index
    const int h = threadIdx.x;
    const int b = r >> 12, i = (r >> 6) & 63, j = r & 63;
    const float e0 = ev[(size_t)r * 3], e1 = ev[(size_t)r * 3 + 1], e2 = ev[(size_t)r * 3 + 2];
    const float eav = ea[r];

    const float*  vi = v  + ((size_t)(b * NA + i) * 3) * HID + h;
    const half_t* qj = vq + ((size_t)(b * NA + j) * 3) * HID + h;
    const half_t* ki = vk + ((size_t)(b * NA + i) * 3) * HID + h;

    const float d2 = vi[0] * e0 + vi[HID] * e1 + vi[2 * HID] * e2;
    const float d3 = (float)qj[0] * (float)ki[0]
                   + (float)qj[HID] * (float)ki[HID]
                   + (float)qj[2 * HID] * (float)ki[2 * HID];

    dirf[(size_t)r * DIR + h]        = (half_t)(d2 * eav);
    dirf[(size_t)r * DIR + HID + h]  = (half_t)(d3 * eav);
}

// ---------------------------------------------------------------------------
// readout: out[b] = (sum_i [z!=0] s[b,i,:]) . Wo + bo
// ---------------------------------------------------------------------------
__global__ __launch_bounds__(128) void readout_k(float* __restrict__ out,
                                                 const float* __restrict__ s,
                                                 const int* __restrict__ z,
                                                 const float* __restrict__ Wo,
                                                 const float* __restrict__ bo) {
    const int b = blockIdx.x;
    const int h = threadIdx.x;
    float acc = 0.f;
    for (int i = 0; i < NA; ++i)
        if (z[b * NA + i] != 0) acc += s[((size_t)(b * NA + i)) * HID + h];
    float val = acc * Wo[h];
    __shared__ float red[HID];
    red[h] = val; __syncthreads();
    for (int st = 64; st > 0; st >>= 1) { if (h < st) red[h] += red[h + st]; __syncthreads(); }
    if (h == 0) out[b] = red[0] + bo[0];   // Y_STD=1, Y_MEAN=0
}

// ---------------------------------------------------------------------------
// host launcher
// ---------------------------------------------------------------------------
extern "C" void kernel_launch(void* const* d_in, const int* in_sizes, int n_in,
                              void* d_out, int out_size, void* d_ws, size_t ws_size,
                              hipStream_t stream) {
    (void)in_sizes; (void)n_in; (void)out_size; (void)ws_size;

    const int*   z      = (const int*)  d_in[0];
    const float* pos    = (const float*)d_in[1];
    const float* emb_w  = (const float*)d_in[2];
    const float* emb2_w = (const float*)d_in[3];
    const float* Wef    = (const float*)d_in[4];
    const float* bef    = (const float*)d_in[5];
    const float* Ws2v   = (const float*)d_in[6];
    const float* bs2v   = (const float*)d_in[7];
    const float* Wq     = (const float*)d_in[8];
    const float* Wk     = (const float*)d_in[9];
    const float* Wd1    = (const float*)d_in[10];
    const float* bd1    = (const float*)d_in[11];
    const float* Wd2    = (const float*)d_in[12];
    const float* bd2    = (const float*)d_in[13];
    const float* Wint   = (const float*)d_in[14];
    const float* bint   = (const float*)d_in[15];
    const float* Wo     = (const float*)d_in[16];
    const float* bo     = (const float*)d_in[17];
    float* out = (float*)d_out;

    // workspace carve
    char* p = (char*)d_ws;
    auto take = [&](size_t bytes) { char* q = p; p += (bytes + 255) & ~(size_t)255; return q; };
    float*  mask = (float*) take((size_t)NP * HID * 4);     // 134 MB
    half_t* dirf = (half_t*)take((size_t)NP * DIR * 2);     // 134 MB
    half_t* h1   = (half_t*)take((size_t)NP * DIR * 2);     // 134 MB
    float*  ea   = (float*) take((size_t)NP * 4);
    float*  ev   = (float*) take((size_t)NP * 3 * 4);
    float*  s    = (float*) take((size_t)ROWS * HID * 4);
    half_t* s_h  = (half_t*)take((size_t)ROWS * HID * 2);
    float*  sn   = (float*) take((size_t)ROWS * HID * 4);
    float*  tmp  = (float*) take((size_t)ROWS * HID * 4);
    float*  t    = (float*) take((size_t)ROWS * HID * 4);
    float*  sl   = (float*) take((size_t)ROWS * HID * 4);
    float*  v    = (float*) take((size_t)ROWS * 3 * HID * 4);
    half_t* v_h  = (half_t*)take((size_t)ROWS * 3 * HID * 2);
    half_t* vq_h = (half_t*)take((size_t)ROWS * 3 * HID * 2);
    half_t* vk_h = (half_t*)take((size_t)ROWS * 3 * HID * 2);
    half_t* Wef_p  = (half_t*)take((size_t)EF * HID * 2);
    half_t* Ws2v_p = (half_t*)take((size_t)HID * HID * 2);
    half_t* Wq_p   = (half_t*)take((size_t)HID * HID * 2);
    half_t* Wk_p   = (half_t*)take((size_t)HID * HID * 2);
    half_t* Wd1_p  = (half_t*)take((size_t)DIR * DIR * 2);
    half_t* Wd2_p  = (half_t*)take((size_t)DIR * HID * 2);
    half_t* Wint_p = (half_t*)take((size_t)3 * HID * HID * 2);

    // 0) pack weights into WMMA fragment order (tiny, once per launch)
    pack_w_k<<<(EF * HID + 255) / 256, 256, 0, stream>>>(Wef_p, Wef, EF, HID);
    pack_w_k<<<(HID * HID + 255) / 256, 256, 0, stream>>>(Ws2v_p, Ws2v, HID, HID);
    pack_w_k<<<(HID * HID + 255) / 256, 256, 0, stream>>>(Wq_p, Wq, HID, HID);
    pack_w_k<<<(HID * HID + 255) / 256, 256, 0, stream>>>(Wk_p, Wk, HID, HID);
    pack_w_k<<<(DIR * DIR + 255) / 256, 256, 0, stream>>>(Wd1_p, Wd1, DIR, DIR);
    pack_w_k<<<(DIR * HID + 255) / 256, 256, 0, stream>>>(Wd2_p, Wd2, DIR, HID);
    for (int l = 0; l < 3; ++l)
        pack_w_k<<<(HID * HID + 255) / 256, 256, 0, stream>>>(
            Wint_p + (size_t)l * HID * HID, Wint + (size_t)l * HID * HID, HID, HID);

    // 1) embeddings + LN(emb2[z])
    embed_k<<<ROWS, 128, 0, stream>>>(s, s_h, sn, z, emb_w, emb2_w);

    // 2) geometry + RBF + ef@Wef (WMMA) -> mask, ea, ev
    ef_mask_k<<<NP / 16, 64, 0, stream>>>(z, pos, Wef_p, bef, mask, ea, ev);

    // 3) NeighborEmb: s += sum_j mask * sn
    agg_k<<<ROWS, 128, 0, stream>>>(s, s_h, mask, sn);

    // 4) t = silu(LN(s @ Ws2v + b))   [4096x128x128]
    gemm_wmma_k<128, 128, false, 0><<<(ROWS / 16) * 2 / 8, 256, 0, stream>>>(
        s_h, Ws2v_p, bs2v, tmp, ROWS);
    ln_k<true><<<ROWS, 128, 0, stream>>>(tmp, t);

    // 5) v = sum_j mask * t * ev   (+f16 shadow)
    cfconv_v_k<<<ROWS, 128, 0, stream>>>(v, v_h, mask, t, ev);

    // 6) vq = v @ Wq, vk = v @ Wk    [12288x128x128, f16 out]
    gemm_wmma_k<128, 128, false, 1><<<(ROWS * 3 / 16) * 2 / 8, 256, 0, stream>>>(
        v_h, Wq_p, nullptr, vq_h, ROWS * 3);
    gemm_wmma_k<128, 128, false, 1><<<(ROWS * 3 / 16) * 2 / 8, 256, 0, stream>>>(
        v_h, Wk_p, nullptr, vk_h, ROWS * 3);

    // 7) directional features (f16) scaled by ea
    dirfeat_k<<<NP, 128, 0, stream>>>(dirf, v, vq_h, vk_h, ev, ea);

    // 8) h1 = silu(dirf @ Wd1 + bd1)   [262144 x 256 x 256, f16 out]
    gemm_wmma_k<256, 256, true, 1><<<(NP / 16) * 4 / 8, 256, 0, stream>>>(
        dirf, Wd1_p, bd1, h1, NP);

    // 9) mask *= silu(h1 @ Wd2 + bd2)  [262144 x 256 x 128, fused dir_proj]
    gemm_wmma_k<256, 128, true, 2><<<(NP / 16) * 2 / 8, 256, 0, stream>>>(
        h1, Wd2_p, bd2, mask, NP);

    // 10) three DirCFConv message-passing layers
    for (int l = 0; l < 3; ++l) {
        gemm_wmma_k<128, 128, false, 0><<<(ROWS / 16) * 2 / 8, 256, 0, stream>>>(
            s_h, Wint_p + (size_t)l * HID * HID, bint + (size_t)l * HID, tmp, ROWS);
        ln_k<true><<<ROWS, 128, 0, stream>>>(tmp, sl);
        agg_k<<<ROWS, 128, 0, stream>>>(s, s_h, mask, sl);
    }

    // 11) readout
    readout_k<<<BB, 128, 0, stream>>>(out, s, z, Wo, bo);
}
